// ShellConvolutionLayer_66022237274250
// MI455X (gfx1250) — matrix operand order repaired and checked by reading, not compiled, not run on hardware
//
#include <hip/hip_runtime.h>

// ---------------------------------------------------------------- constants
#define NNODES 100000
#define NEDGES 640000
#define DIN    128          // atom_input_dim
#define DOUT   256          // output_dim
#define KIN    512          // (HOPS+1)*DIN
#define HOPS   3

typedef __attribute__((ext_vector_type(16))) __bf16 bf16x16;
typedef __attribute__((ext_vector_type(8)))  __bf16 bf16x8;
typedef __attribute__((ext_vector_type(8)))  float  f32x8;

// ------------------------------------------------ CDNA5 async LDS staging
// GLOBAL_LOAD_ASYNC_TO_LDS_B128: per-lane 16B global->LDS copy, ASYNCcnt.
__device__ __forceinline__ void async_copy_b128(const void* gsrc, void* lds_dst) {
    unsigned lds = (unsigned)(size_t)lds_dst;      // addr[31:0] == LDS offset
    asm volatile("global_load_async_to_lds_b128 %0, %1, off"
                 :: "v"(lds), "v"(gsrc) : "memory");
}
__device__ __forceinline__ void async_wait0() {
    asm volatile("s_wait_asynccnt 0" ::: "memory");
}

// ---------------------------------------------------------------- utilities
__global__ void zero_f32x4_kernel(float4* __restrict__ p, long long n4) {
    long long i = (long long)blockIdx.x * blockDim.x + threadIdx.x;
    long long stride = (long long)gridDim.x * blockDim.x;
    float4 z; z.x = z.y = z.z = z.w = 0.f;
    for (; i < n4; i += stride) p[i] = z;
}

// agg[target[e]] += x[src[e] % N]   (expanded = tile(x,3) so expanded[s] = x[s%N])
__global__ void scatter_kernel(const float* __restrict__ x,
                               const int* __restrict__ target,
                               const int* __restrict__ src,
                               float* __restrict__ agg) {
    int tid = blockIdx.x * blockDim.x + threadIdx.x;   // NEDGES*32 threads
    int e = tid >> 5;
    if (e >= NEDGES) return;
    int q = (tid & 31) * 4;                            // 4 floats per thread
    int s = src[e];
    int srcn = s % NNODES;
    int tg = target[e];
    float4 v = *(const float4*)(x + (size_t)srcn * DIN + q);
    float* dst = agg + (size_t)tg * DIN + q;
    atomicAdd(dst + 0, v.x);
    atomicAdd(dst + 1, v.y);
    atomicAdd(dst + 2, v.z);
    atomicAdd(dst + 3, v.w);
}

// Build n-major (transposed) bf16 weights:
//   WT1[n][k] = n<256 ? W_in[k][n] : W_skip[k][n-256]      (512 x 512)
//   WTb[w][n][k] = Wblk[w][k][n]   for W1_0,W2_0,W1_1,W2_1 (256 x 256 each)
__global__ void convert_weights_kernel(const float* __restrict__ Win,
                                       const float* __restrict__ Wskip,
                                       const float* __restrict__ W10,
                                       const float* __restrict__ W20,
                                       const float* __restrict__ W11,
                                       const float* __restrict__ W21,
                                       __bf16* __restrict__ WT1,
                                       __bf16* __restrict__ WTb) {
    int i = blockIdx.x * blockDim.x + threadIdx.x;
    if (i < 512 * 512) {
        int n = i >> 9, k = i & 511;
        float v = (n < 256) ? Win[(size_t)k * 256 + n] : Wskip[(size_t)k * 256 + (n - 256)];
        WT1[i] = (__bf16)v;
    } else {
        int j = i - 512 * 512;
        if (j >= 4 * 256 * 256) return;
        int w = j >> 16, r = j & 0xFFFF;
        int n = r >> 8, k = r & 255;
        const float* Ws = (w == 0) ? W10 : (w == 1) ? W20 : (w == 2) ? W11 : W21;
        WTb[(size_t)w * 65536 + (size_t)n * 256 + k] = (__bf16)Ws[(size_t)k * 256 + n];
    }
}

__device__ __forceinline__ float silu_f(float v) {
    return v * (1.f / (1.f + __expf(-v)));
}
__device__ __forceinline__ __bf16 to_bf16(float v) { return (__bf16)v; }

// ---------------------------------------------------------------- GEMM 1
// Block tile 128M x 64N, K-step 32, double-buffered LDS pipeline.
__launch_bounds__(128)
__global__ void gemm1_kernel(const float* __restrict__ x,
                             const float* __restrict__ agg,
                             const __bf16* __restrict__ WT1,
                             const float* __restrict__ b_in,
                             const float* __restrict__ b_skip,
                             float* __restrict__ h,
                             __bf16* __restrict__ hb,
                             float* __restrict__ skip) {
    __shared__ alignas(16) __bf16 As[2][128][40];   // 80B pitch: conflict-free
    __shared__ alignas(16) __bf16 Bs[2][64][40];

    const int m0 = blockIdx.x * 128;
    const int n0 = blockIdx.y * 64;
    const int t = threadIdx.x;
    const int lane = t & 31, wv = t >> 5;
    const int hi = lane >> 4, ln = lane & 15;
    const int rb = t >> 1, part = t & 1;            // B-stage mapping

    f32x8 acc[2][4] = {};

    const int rowA = m0 + t;
    const bool rowOk = rowA < NNODES;

    auto stageA = [&](int kks, int buf) {
        int seg = kks >> 7, off = kks & 127;        // 32-chunk stays in one 128-seg
        const float* sp = (seg == 0)
            ? (x + (size_t)rowA * DIN + off)
            : (agg + ((size_t)(seg - 1) * NNODES + rowA) * DIN + off);
        if (rowOk) {
#pragma unroll
            for (int j = 0; j < 4; ++j) {           // 8 f32 -> 8 bf16 (16B)
                float4 v0 = *(const float4*)(sp + j * 8);
                float4 v1 = *(const float4*)(sp + j * 8 + 4);
                bf16x8 c;
                c[0] = to_bf16(v0.x); c[1] = to_bf16(v0.y);
                c[2] = to_bf16(v0.z); c[3] = to_bf16(v0.w);
                c[4] = to_bf16(v1.x); c[5] = to_bf16(v1.y);
                c[6] = to_bf16(v1.z); c[7] = to_bf16(v1.w);
                *(bf16x8*)&As[buf][t][j * 8] = c;
            }
        } else {
            bf16x8 z = {};
#pragma unroll
            for (int j = 0; j < 4; ++j) *(bf16x8*)&As[buf][t][j * 8] = z;
        }
    };
    auto stageB = [&](int kks, int buf) {
        const __bf16* bp = WT1 + (size_t)(n0 + rb) * KIN + kks + part * 16;
        async_copy_b128(bp,     &Bs[buf][rb][part * 16]);
        async_copy_b128(bp + 8, &Bs[buf][rb][part * 16 + 8]);
    };
    auto compute = [&](int buf) {
        bf16x16 af[2];
#pragma unroll
        for (int sub = 0; sub < 2; ++sub) {
            bf16x8 alo = *(const bf16x8*)&As[buf][wv * 32 + sub * 16 + ln][hi * 16];
            bf16x8 ahi = *(const bf16x8*)&As[buf][wv * 32 + sub * 16 + ln][hi * 16 + 8];
            af[sub] = __builtin_shufflevector(alo, ahi,
                0, 1, 2, 3, 4, 5, 6, 7, 8, 9, 10, 11, 12, 13, 14, 15);
        }
#pragma unroll
        for (int nt = 0; nt < 4; ++nt) {
            bf16x16 b = *(const bf16x16*)&Bs[buf][nt * 16 + ln][hi * 16];
            acc[0][nt] = __builtin_amdgcn_wmma_f32_16x16x32_bf16(
                false, af[0], false, b, (short)0, acc[0][nt], false, false);
            acc[1][nt] = __builtin_amdgcn_wmma_f32_16x16x32_bf16(
                false, af[1], false, b, (short)0, acc[1][nt], false, false);
        }
    };

    // pipeline: async copies for `nxt` overlap WMMA on `cur`; 1 barrier/step
    stageA(0, 0);
    stageB(0, 0);
    int cur = 0;
    for (int kk = 0; kk < KIN; kk += 32) {
        async_wait0();                 // only `cur` copies outstanding here
        __syncthreads();
        int nxt = cur ^ 1;
        if (kk + 32 < KIN) { stageA(kk + 32, nxt); stageB(kk + 32, nxt); }
        compute(cur);
        cur = nxt;
    }

    const bool isH = (n0 < DOUT);
#pragma unroll
    for (int sub = 0; sub < 2; ++sub) {
#pragma unroll
        for (int nt = 0; nt < 4; ++nt) {
            int col = n0 + nt * 16 + ln;
            float bias = isH ? b_in[col] : b_skip[col - DOUT];
#pragma unroll
            for (int r = 0; r < 8; ++r) {
                int row = m0 + wv * 32 + sub * 16 + hi * 8 + r;
                if (row >= NNODES) continue;
                float v = acc[sub][nt][r] + bias;
                if (isH) {
                    v = silu_f(v);
                    h [(size_t)row * DOUT + col] = v;
                    hb[(size_t)row * DOUT + col] = to_bf16(v);
                } else {
                    skip[(size_t)row * DOUT + (col - DOUT)] = v;
                }
            }
        }
    }
}

// ---------------------------------------------------------------- GEMM 2
// A bf16 [N][256]: A and B tiles staged fully async, double-buffered.
template <bool SILU>
__launch_bounds__(128)
__global__ void gemm2_kernel(const __bf16* __restrict__ Abf,
                             const __bf16* __restrict__ WT,
                             const float* __restrict__ bias,
                             const float* __restrict__ res1,
                             const float* __restrict__ res2,
                             float* __restrict__ dst_f32,
                             __bf16* __restrict__ dst_bf16) {
    __shared__ alignas(16) __bf16 As[2][128][40];
    __shared__ alignas(16) __bf16 Bs[2][64][40];

    const int m0 = blockIdx.x * 128;
    const int n0 = blockIdx.y * 64;
    const int t = threadIdx.x;
    const int lane = t & 31, wv = t >> 5;
    const int hi = lane >> 4, ln = lane & 15;
    const int rb = t >> 1, part = t & 1;

    f32x8 acc[2][4] = {};

    const int rowA = m0 + t;
    const bool rowOk = rowA < NNODES;

    auto stageA = [&](int kks, int buf) {
        if (rowOk) {
            const __bf16* sp = Abf + (size_t)rowA * DOUT + kks;
#pragma unroll
            for (int j = 0; j < 4; ++j)
                async_copy_b128(sp + j * 8, &As[buf][t][j * 8]);
        } else {
            bf16x8 z = {};
#pragma unroll
            for (int j = 0; j < 4; ++j) *(bf16x8*)&As[buf][t][j * 8] = z;
        }
    };
    auto stageB = [&](int kks, int buf) {
        const __bf16* bp = WT + (size_t)(n0 + rb) * DOUT + kks + part * 16;
        async_copy_b128(bp,     &Bs[buf][rb][part * 16]);
        async_copy_b128(bp + 8, &Bs[buf][rb][part * 16 + 8]);
    };
    auto compute = [&](int buf) {
        bf16x16 af[2];
#pragma unroll
        for (int sub = 0; sub < 2; ++sub) {
            bf16x8 alo = *(const bf16x8*)&As[buf][wv * 32 + sub * 16 + ln][hi * 16];
            bf16x8 ahi = *(const bf16x8*)&As[buf][wv * 32 + sub * 16 + ln][hi * 16 + 8];
            af[sub] = __builtin_shufflevector(alo, ahi,
                0, 1, 2, 3, 4, 5, 6, 7, 8, 9, 10, 11, 12, 13, 14, 15);
        }
#pragma unroll
        for (int nt = 0; nt < 4; ++nt) {
            bf16x16 b = *(const bf16x16*)&Bs[buf][nt * 16 + ln][hi * 16];
            acc[0][nt] = __builtin_amdgcn_wmma_f32_16x16x32_bf16(
                false, af[0], false, b, (short)0, acc[0][nt], false, false);
            acc[1][nt] = __builtin_amdgcn_wmma_f32_16x16x32_bf16(
                false, af[1], false, b, (short)0, acc[1][nt], false, false);
        }
    };

    stageA(0, 0);
    stageB(0, 0);
    int cur = 0;
    for (int kk = 0; kk < DOUT; kk += 32) {
        async_wait0();
        __syncthreads();
        int nxt = cur ^ 1;
        if (kk + 32 < DOUT) { stageA(kk + 32, nxt); stageB(kk + 32, nxt); }
        compute(cur);
        cur = nxt;
    }

#pragma unroll
    for (int sub = 0; sub < 2; ++sub) {
#pragma unroll
        for (int nt = 0; nt < 4; ++nt) {
            int col = n0 + nt * 16 + ln;
            float bv = bias[col];
#pragma unroll
            for (int r = 0; r < 8; ++r) {
                int row = m0 + wv * 32 + sub * 16 + hi * 8 + r;
                if (row >= NNODES) continue;
                float v = acc[sub][nt][r] + bv;
                if (SILU) v = silu_f(v);
                size_t idx = (size_t)row * DOUT + col;
                if (res1) v += res1[idx];
                if (res2) v += res2[idx];
                if (dst_f32)  dst_f32[idx] = v;
                if (dst_bf16) dst_bf16[idx] = to_bf16(v);
            }
        }
    }
}

// ---------------------------------------------------------------- launch
extern "C" void kernel_launch(void* const* d_in, const int* in_sizes, int n_in,
                              void* d_out, int out_size, void* d_ws, size_t ws_size,
                              hipStream_t stream) {
    const float* x      = (const float*)d_in[0];
    const int*   target = (const int*)d_in[1];
    const int*   src    = (const int*)d_in[2];
    const float* W_in   = (const float*)d_in[3];
    const float* b_in   = (const float*)d_in[4];
    const float* W1_0   = (const float*)d_in[5];
    const float* b1_0   = (const float*)d_in[6];
    const float* W2_0   = (const float*)d_in[7];
    const float* b2_0   = (const float*)d_in[8];
    const float* W1_1   = (const float*)d_in[9];
    const float* b1_1   = (const float*)d_in[10];
    const float* W2_1   = (const float*)d_in[11];
    const float* b2_1   = (const float*)d_in[12];
    const float* W_skip = (const float*)d_in[13];
    const float* b_skip = (const float*)d_in[14];
    float* out = (float*)d_out;

    // workspace carve-up (~359 MB)
    char* ws = (char*)d_ws;
    size_t off = 0;
    float*  agg = (float*)(ws + off);  off += (size_t)HOPS * NNODES * DIN * sizeof(float);  // 153.6 MB
    float*  h   = (float*)(ws + off);  off += (size_t)NNODES * DOUT * sizeof(float);        // 102.4 MB
    __bf16* hb  = (__bf16*)(ws + off); off += (size_t)NNODES * DOUT * sizeof(__bf16);       //  51.2 MB
    __bf16* tbb = (__bf16*)(ws + off); off += (size_t)NNODES * DOUT * sizeof(__bf16);       //  51.2 MB
    __bf16* WT1 = (__bf16*)(ws + off); off += (size_t)KIN * KIN * sizeof(__bf16);           // 512 KB
    __bf16* WTb = (__bf16*)(ws + off); off += (size_t)4 * DOUT * DOUT * sizeof(__bf16);     // 512 KB
    (void)ws_size; (void)in_sizes; (void)n_in; (void)out_size;

    // 1) zero agg (atomics accumulate, must re-zero each call)
    {
        long long n4 = (long long)HOPS * NNODES * DIN / 4;
        zero_f32x4_kernel<<<4096, 256, 0, stream>>>((float4*)agg, n4);
    }
    // 2) edge scatter-add
    {
        long long threads = (long long)NEDGES * 32;
        int blocks = (int)((threads + 255) / 256);
        scatter_kernel<<<blocks, 256, 0, stream>>>(x, target, src, agg);
    }
    // 3) bf16 transposed weights
    {
        int total = 512 * 512 + 4 * 256 * 256;
        convert_weights_kernel<<<(total + 255) / 256, 256, 0, stream>>>(
            W_in, W_skip, W1_0, W2_0, W1_1, W2_1, WT1, WTb);
    }
    // 4) fused input GEMM: h(+hb) = silu(feats@W_in+b_in); d_out = feats@W_skip+b_skip
    {
        dim3 grid((NNODES + 127) / 128, KIN / 64);
        gemm1_kernel<<<grid, 128, 0, stream>>>(x, agg, WT1, b_in, b_skip, h, hb, out);
    }
    // 5) residual MLP blocks
    {
        dim3 grid((NNODES + 127) / 128, DOUT / 64);
        const __bf16* W10T = WTb + (size_t)0 * 65536;
        const __bf16* W20T = WTb + (size_t)1 * 65536;
        const __bf16* W11T = WTb + (size_t)2 * 65536;
        const __bf16* W21T = WTb + (size_t)3 * 65536;
        // block 0
        gemm2_kernel<true ><<<grid, 128, 0, stream>>>(hb,  W10T, b1_0, nullptr, nullptr, nullptr, tbb);
        gemm2_kernel<false><<<grid, 128, 0, stream>>>(tbb, W20T, b2_0, h,       nullptr, h,       hb);
        // block 1 (final fuses +h residual and +skip already in d_out)
        gemm2_kernel<true ><<<grid, 128, 0, stream>>>(hb,  W11T, b1_1, nullptr, nullptr, nullptr, tbb);
        gemm2_kernel<false><<<grid, 128, 0, stream>>>(tbb, W21T, b2_1, h,       out,     out,     nullptr);
    }
}
